// DeformConv2d_sphe_27779848471006
// MI455X (gfx1250) — compile-verified, matrix operand order checked by reading, not run on hardware
//
#include <hip/hip_runtime.h>

typedef __attribute__((ext_vector_type(16))) _Float16 v16h;
typedef __attribute__((ext_vector_type(8)))  float    v8f;

// Problem constants (match reference setup_inputs)
constexpr int BB   = 8;
constexpr int CIN  = 64;
constexpr int Hh   = 96;
constexpr int Ww   = 96;
constexpr int COUT = 128;
constexpr int KTAP = 9;            // 3x3
constexpr int Ho   = 96;
constexpr int Wo   = 96;
constexpr int HW   = Ho * Wo;      // 9216
constexpr int KK   = CIN * KTAP;   // 576  (GEMM K)
constexpr int KC   = KK / 32;      // 18   (k-chunks of 32)
constexpr int NT   = COUT / 16;    // 8    (cout 16-tiles)
constexpr int MT   = 32;           // output positions per block (2 x 16-row subtiles)
constexpr int SPT  = MT * KTAP;    // 288 sample points per block

// --- map k_local (0..31) inside a 32-k chunk to (laneHi, half-index hh) per the
// ISA 16-bit 32x16 B-matrix layout:  k_local = 2v + p + 8*(v>=4) + 8*laneHi
__device__ __forceinline__ void klocal_to_slot(int kl, int& laneHi, int& hh) {
    laneHi = (kl >> 3) & 1;
    int kp = kl & ~8;                 // in {0..7} U {16..23}
    int v  = (kp < 8) ? (kp >> 1) : (4 + ((kp - 16) >> 1));
    int p  = kp & 1;
    hh = 2 * v + p;
}

// Repack f32 weights [Cout,Cin,3,3] -> f16 B fragments:
// wp[nt][kc][lane][hh], 16 halves (32B) contiguous per lane.
__global__ void dcn_repack_weights(const float* __restrict__ w,
                                   _Float16* __restrict__ wp) {
    int idx = blockIdx.x * 256 + threadIdx.x;     // total NT*KC*32*16 = 73728
    if (idx >= NT * KC * 32 * 16) return;
    int hh   = idx & 15;
    int lane = (idx >> 4) & 31;
    int rest = idx >> 9;
    int kc   = rest % KC;
    int nt   = rest / KC;
    int n = nt * 16 + (lane & 15);
    int v = hh >> 1, p = hh & 1;
    int kl = 2 * v + p + (v >= 4 ? 8 : 0) + ((lane >= 16) ? 8 : 0);
    int k  = kc * 32 + kl;            // k = c*9 + tap
    int c = k / KTAP, tap = k % KTAP;
    wp[idx] = (_Float16)w[(n * CIN + c) * KTAP + tap];
}

__device__ __forceinline__ v8f wmma_f16(v16h a, v16h b, v8f c) {
    return __builtin_amdgcn_wmma_f32_16x16x32_f16(
        /*neg_a=*/false, a, /*neg_b=*/false, b,
        /*c_mod=*/(short)0, c, /*reuse_a=*/false, /*reuse_b=*/false);
}

__global__ __launch_bounds__(256)
void dcn_wmma_kernel(const float* __restrict__ x,
                     const float* __restrict__ offset,
                     const float* __restrict__ mask,
                     const _Float16* __restrict__ wpack,
                     const float* __restrict__ bias,
                     float* __restrict__ out) {
    // A fragments, WMMA lane layout: Afrag[sub(2)][kc(18)][lane(32)][hh(16)]
    __shared__ __align__(32) _Float16 Afrag[2 * KC * 32 * 16];   // 36 KB
    __shared__ int   sIdx[SPT][4];
    __shared__ float sW  [SPT][4];

    const int tid = threadIdx.x;
    const int mbase   = blockIdx.x * MT;  // HW % 32 == 0 -> no batch straddle
    const int b       = mbase / HW;
    const int sp_base = mbase % HW;

    // ---- stage 1: bilinear corner weights/indices for 32 pos x 9 taps
    for (int s = tid; s < SPT; s += 256) {
        int pos = s / KTAP;
        int tap = s % KTAP;
        int sp  = sp_base + pos;
        int oh  = sp / Wo, ow = sp % Wo;
        int ky  = tap / 3, kx = tap % 3;
        float dy = offset[(b * 2 * KTAP + tap * 2 + 0) * HW + sp];
        float dx = offset[(b * 2 * KTAP + tap * 2 + 1) * HW + sp];
        float m  = mask[(b * KTAP + tap) * HW + sp];
        float py = (float)(oh - 1 + ky) + dy;
        float px = (float)(ow - 1 + kx) + dx;
        float y0 = floorf(py), x0 = floorf(px);
        float fy = py - y0,   fx = px - x0;
        float wy[2] = {1.f - fy, fy};
        float wx[2] = {1.f - fx, fx};
#pragma unroll
        for (int j = 0; j < 4; ++j) {
            int ddy = j >> 1, ddx = j & 1;
            float yc = y0 + (float)ddy, xc = x0 + (float)ddx;
            bool valid = (yc >= 0.f) && (yc < (float)Hh) &&
                         (xc >= 0.f) && (xc < (float)Ww);
            int yi = (int)fminf(fmaxf(yc, 0.f), (float)(Hh - 1));
            int xi = (int)fminf(fmaxf(xc, 0.f), (float)(Ww - 1));
            sIdx[s][j] = yi * Ww + xi;
            sW[s][j]   = valid ? (wy[ddy] * wx[ddx] * m) : 0.f;
        }
    }
    __syncthreads();

    // ---- stage 2: gather 32*9*64 values, scatter into A-fragment layout
    const float* xb = x + (size_t)b * CIN * Hh * Ww;
    for (int e = tid; e < SPT * CIN; e += 256) {
        int pt = e / CIN;                 // pos*9 + tap
        int c  = e % CIN;
        int pos = pt / KTAP, tap = pt % KTAP;
        const float* xc = xb + (size_t)c * (Hh * Ww);
        float v = sW[pt][0] * xc[sIdx[pt][0]] + sW[pt][1] * xc[sIdx[pt][1]] +
                  sW[pt][2] * xc[sIdx[pt][2]] + sW[pt][3] * xc[sIdx[pt][3]];
        int k  = c * KTAP + tap;          // GEMM k index (c-major, matches repack)
        int kc = k >> 5;
        int laneHi, hh;
        klocal_to_slot(k & 31, laneHi, hh);
        int sub  = pos >> 4;
        int lane = laneHi * 16 + (pos & 15);
        Afrag[((sub * KC + kc) * 32 + lane) * 16 + hh] = (_Float16)v;
    }
    __syncthreads();

    // ---- stage 3: waves arranged 2(M) x 4(N); each wave: 16 rows x 32 couts,
    // 2 WMMA per k-chunk sharing one A fragment; software-pipelined loads.
    const int wave = tid >> 5;
    const int lane = tid & 31;
    const int mi   = wave >> 2;           // M subtile 0..1
    const int ng   = wave & 3;            // cout group: couts [ng*32, ng*32+32)
    v8f acc0 = {}, acc1 = {};
    const v16h* aptr  = (const v16h*)Afrag + (size_t)(mi * KC) * 32 + lane;
    const v16h* bptr0 = (const v16h*)wpack + (size_t)((ng * 2 + 0) * KC) * 32 + lane;
    const v16h* bptr1 = (const v16h*)wpack + (size_t)((ng * 2 + 1) * KC) * 32 + lane;

    v16h a  = aptr[0];
    v16h b0 = bptr0[0];
    v16h b1 = bptr1[0];
#pragma unroll 1
    for (int kc = 0; kc < KC; ++kc) {
        int nk = (kc + 1 < KC) ? (kc + 1) : kc;   // keep loads in-bounds
        __builtin_prefetch((const void*)(bptr0 + nk * 32), 0, 1);
        v16h an  = aptr[nk * 32];
        v16h b0n = bptr0[nk * 32];
        v16h b1n = bptr1[nk * 32];
        acc0 = wmma_f16(a, b0, acc0);
        acc1 = wmma_f16(a, b1, acc1);
        a = an; b0 = b0n; b1 = b1n;
    }

    // ---- write out per 16x16 f32 C/D layout (VGPR r: M=r lanes0-15, M=8+r hi)
    int n0 = (ng * 2 + 0) * 16 + (lane & 15);
    int n1 = (ng * 2 + 1) * 16 + (lane & 15);
    float bv0 = bias[n0];
    float bv1 = bias[n1];
    int mrow = (lane < 16) ? 0 : 8;
    int sp0  = sp_base + mi * 16 + mrow;
    float* o0 = out + (size_t)(b * COUT + n0) * HW + sp0;
    float* o1 = out + (size_t)(b * COUT + n1) * HW + sp0;
#pragma unroll
    for (int r = 0; r < 8; ++r) {
        o0[r] = acc0[r] + bv0;
        o1[r] = acc1[r] + bv1;
    }
}

extern "C" void kernel_launch(void* const* d_in, const int* in_sizes, int n_in,
                              void* d_out, int out_size, void* d_ws, size_t ws_size,
                              hipStream_t stream) {
    const float* x      = (const float*)d_in[0];
    const float* offset = (const float*)d_in[1];
    const float* mask   = (const float*)d_in[2];
    const float* weight = (const float*)d_in[3];
    const float* bias   = (const float*)d_in[4];
    float* out          = (float*)d_out;
    _Float16* wpack     = (_Float16*)d_ws;     // 73,728 halves = 147,456 B

    // Repack weights into WMMA B-fragment layout (re-done every call: stateless).
    dcn_repack_weights<<<(NT * KC * 32 * 16 + 255) / 256, 256, 0, stream>>>(weight, wpack);

    // One block per 32 output positions: 73728/32 = 2304 blocks, 8 waves each.
    const int mtiles = (BB * HW) / MT;
    dcn_wmma_kernel<<<mtiles, 256, 0, stream>>>(x, offset, mask, wpack, bias, out);
}
